// HyperTokenMixer_44616120271558
// MI455X (gfx1250) — matrix operand order repaired and checked by reading, not compiled, not run on hardware
//
#include <hip/hip_runtime.h>
#include <hip/hip_bf16.h>
#include <stdint.h>

// ---------------------------------------------------------------------------
// HyperTokenMixer on gfx1250 (MI455X): bf16 WMMA GEMMs, f32 accumulate,
// double-buffered LDS pipeline, TDM (tensor_load_to_lds) for row-major tiles.
//   Xp = X + P                       [b,n,d]
//   W1 = Xp@Wh1 + bh1, W2 = Xp@Wh2 + bh2  [b,n,dp]
//   Pm = W1^T @ Xp  (per batch)      [b,dp,d]
//   A  = gelu_exact(Pm)              [b,dp,d]
//   Y  = W2 @ A     (per batch)      [b,n,d]
// ---------------------------------------------------------------------------

typedef __attribute__((ext_vector_type(16))) __bf16 v16bf;
typedef __attribute__((ext_vector_type(8)))  float  v8f;
typedef __attribute__((ext_vector_type(4)))  unsigned int u32x4;
typedef __attribute__((ext_vector_type(8)))  int          i32x8;
typedef __attribute__((ext_vector_type(4)))  int          i32x4;

#define BATCH 8
#define NTOK  4096
#define DIM   256
#define DPRM  512

#define BM 128
#define BN 128
#define BK 32
#define LDK 40          // padded LDS row: 40 bf16 = 80 B = 20 dwords -> 16B-aligned rows
#define THREADS 256     // 8 waves of 32
#define KSPLIT 4

typedef unsigned short bf16s;   // bf16 bits in memory
typedef unsigned int   u32;

#if defined(__has_builtin)
#if __has_builtin(__builtin_amdgcn_tensor_load_to_lds) && \
    __has_builtin(__builtin_amdgcn_s_wait_tensorcnt)
#define HAS_TDM 1
#else
#define HAS_TDM 0
#endif
#else
#define HAS_TDM 0
#endif

__device__ __forceinline__ bf16s f2bf(float f) {
    u32 u = __float_as_uint(f);
    u32 r = (u + 0x7FFFu + ((u >> 16) & 1u)) >> 16;   // RNE
    return (bf16s)r;
}

// ---- WMMA fragment loads from LDS (row-major, LDK stride) -----------------
// A 16x32 bf16: lane<16 -> row=lane, K dwords {0..3}+{8..11}; lane>=16 adds 4.
__device__ __forceinline__ v16bf frag_ld_A(const bf16s* tile, int lane) {
    const u32* p = (const u32*)(tile + (lane & 15) * LDK);
    int h4 = (lane >> 4) * 4;
    union { v16bf v; u32 u[8]; } f;
#pragma unroll
    for (int i = 0; i < 8; ++i) {
        int dw = (i < 4 ? i : i + 4) + h4;
        f.u[i] = p[dw];
    }
    return f.v;
}
// B staged N-major (Bt: N rows x K cols): lane<16 -> K dwords {0..3}+{8..11}; lane>=16 adds 4.
__device__ __forceinline__ v16bf frag_ld_B(const bf16s* tile, int lane) {
    const u32* p = (const u32*)(tile + (lane & 15) * LDK);
    int h4 = (lane >> 4) * 4;
    union { v16bf v; u32 u[8]; } f;
#pragma unroll
    for (int i = 0; i < 8; ++i) f.u[i] = p[(i < 4 ? i : i + 4) + h4];
    return f.v;
}

__device__ __forceinline__ v8f wmma_bf16(v16bf a, v16bf b, v8f c) {
    return __builtin_amdgcn_wmma_f32_16x16x32_bf16(false, a, false, b, (short)0, c,
                                                   false, false);
}

// one BK step: each wave does 2 (M) x 4 (N) 16x16 tiles
__device__ __forceinline__ void mma_step(const bf16s* lds_a, const bf16s* lds_b,
                                         int wm, int wn, int lane, v8f acc[2][4]) {
    v16bf af[2], bfq[4];
#pragma unroll
    for (int i = 0; i < 2; ++i)
        af[i] = frag_ld_A(lds_a + (wm * 32 + i * 16) * LDK, lane);
#pragma unroll
    for (int j = 0; j < 4; ++j)
        bfq[j] = frag_ld_B(lds_b + (wn * 64 + j * 16) * LDK, lane);
#pragma unroll
    for (int i = 0; i < 2; ++i)
#pragma unroll
        for (int j = 0; j < 4; ++j)
            acc[i][j] = wmma_bf16(af[i], bfq[j], acc[i][j]);
}

__device__ __forceinline__ void zero_acc(v8f acc[2][4]) {
    v8f z = {0.f, 0.f, 0.f, 0.f, 0.f, 0.f, 0.f, 0.f};
#pragma unroll
    for (int i = 0; i < 2; ++i)
#pragma unroll
        for (int j = 0; j < 4; ++j) acc[i][j] = z;
}

// ---- TDM: async DMA of a 2-D [128 x 32] bf16 tile global->LDS, with the
// +4-dword row padding applied in LDS (pad_interval=16 dw, pad_amount=4 dw)
// so the landed layout matches LDK=40. Issue from one wave; TENSORcnt tracks.
__device__ __forceinline__ void tdm_load_tile(void* lds_dst, const void* gsrc,
                                              unsigned stride_elems) {
#if HAS_TDM
    unsigned long long ga = (unsigned long long)(uintptr_t)gsrc;
    unsigned lo = (unsigned)(uintptr_t)lds_dst;
    u32x4 g0;
    g0[0] = 1u;                                   // count=1, user descriptor
    g0[1] = lo;                                   // lds_addr (bytes)
    g0[2] = (u32)ga;                              // global_addr[31:0]
    g0[3] = (u32)(ga >> 32) | (2u << 30);         // global_addr[56:32] | type=2
    i32x8 g1;
    g1[0] = (1 << 16) | (1 << 20) | (3 << 22) | (3 << 25);
    //       data=2B     pad_en     pad_int=16dw   pad_amt=4dw
    g1[1] = (int)((stride_elems & 0xFFFFu) << 16);   // tensor_dim0[15:0]
    g1[2] = (int)(0xFFFFu << 16);                    // dim0 hi=0 | tensor_dim1 lo
    g1[3] = (int)(32u << 16);                        // dim1 hi=0 | tile_dim0=32
    g1[4] = 128;                                     // tile_dim1=128, tile_dim2=0
    g1[5] = (int)stride_elems;                       // tensor_dim0_stride lo
    g1[6] = 0;
    g1[7] = 0;
    i32x4 z4 = {0, 0, 0, 0};
    i32x8 z8 = {0, 0, 0, 0, 0, 0, 0, 0};
    __builtin_amdgcn_tensor_load_to_lds(g0, g1, z4, z4, z8, 0);
#else
    (void)lds_dst; (void)gsrc; (void)stride_elems;
#endif
}

// ---- manual staging, register-split for software pipelining ---------------
// direct: 128 rows x 32 bf16, rows contiguous in K -> b128 global loads
struct RegsDirect { uint4 q[2]; };
__device__ __forceinline__ void ld_direct(RegsDirect& r, const bf16s* src,
                                          long stride, int tid) {
#pragma unroll
    for (int t = 0; t < 2; ++t) {
        int idx = tid + t * THREADS;     // 0..511
        int row = idx >> 2, q = idx & 3;
        r.q[t] = ((const uint4*)(src + (long)row * stride))[q];
    }
}
__device__ __forceinline__ void st_direct(const RegsDirect& r, bf16s* lds, int tid) {
    u32* dl = (u32*)lds;
#pragma unroll
    for (int t = 0; t < 2; ++t) {
        int idx = tid + t * THREADS;
        int row = idx >> 2, q = idx & 3;
        *(uint4*)(dl + row * (LDK / 2) + q * 4) = r.q[t];   // 16B-aligned
    }
}
// transpose: src [32 k-rows x 128 cols] contiguous in col -> lds[col][k]
struct RegsTrans { uint2 q[4]; };
__device__ __forceinline__ void ld_trans(RegsTrans& r, const bf16s* src,
                                         long stride, int tid) {
#pragma unroll
    for (int t = 0; t < 4; ++t) {
        int idx = tid + t * THREADS;     // 0..1023
        int kk = idx >> 5, c4 = (idx & 31) * 4;
        r.q[t] = *(const uint2*)(src + (long)kk * stride + c4);
    }
}
__device__ __forceinline__ void st_trans(const RegsTrans& r, bf16s* lds, int tid) {
#pragma unroll
    for (int t = 0; t < 4; ++t) {
        int idx = tid + t * THREADS;
        int kk = idx >> 5, c4 = (idx & 31) * 4;
        uint2 v = r.q[t];
        lds[(c4 + 0) * LDK + kk] = (bf16s)(v.x & 0xFFFFu);
        lds[(c4 + 1) * LDK + kk] = (bf16s)(v.x >> 16);
        lds[(c4 + 2) * LDK + kk] = (bf16s)(v.y & 0xFFFFu);
        lds[(c4 + 3) * LDK + kk] = (bf16s)(v.y >> 16);
    }
}

// ---------------------------------------------------------------------------
// Kernel 1: Xp = X + P -> bf16, vectorized x4
__global__ void k_prep_xp(const float* __restrict__ X, const float* __restrict__ P,
                          bf16s* __restrict__ xp, long n4) {
    long i = (long)blockIdx.x * blockDim.x + threadIdx.x;
    long gs = (long)gridDim.x * blockDim.x;
    const float4* x4 = (const float4*)X;
    const float4* p4 = (const float4*)P;
    for (; i < n4; i += gs) {
        float4 a = x4[i], b = p4[i];
        bf16s* o = xp + i * 4;
        o[0] = f2bf(a.x + b.x); o[1] = f2bf(a.y + b.y);
        o[2] = f2bf(a.z + b.z); o[3] = f2bf(a.w + b.w);
    }
}

// Kernel 2: Wh [d,dp] -> WhT [dp,d] bf16 (both weight sets)
__global__ void k_prep_whT(const float* __restrict__ Wh1, const float* __restrict__ Wh2,
                           bf16s* __restrict__ whT1, bf16s* __restrict__ whT2) {
    int idx = blockIdx.x * blockDim.x + threadIdx.x;    // over DIM*DPRM
    if (idx >= DIM * DPRM) return;
    int r = idx / DPRM, c = idx % DPRM;
    whT1[c * DIM + r] = f2bf(Wh1[idx]);
    whT2[c * DIM + r] = f2bf(Wh2[idx]);
}

// Kernel 3: W{1,2}[b*n, dp] = Xp[b*n, d] @ Wh + bh  (grid.z: 0->W1, 1->W2)
// Both tiles row-major => TDM double-buffered staging, zero VMEM in loop.
__global__ void __launch_bounds__(THREADS)
k_hyper_w(const bf16s* __restrict__ xp,
          const bf16s* __restrict__ whT1, const bf16s* __restrict__ whT2,
          const float* __restrict__ bh1, const float* __restrict__ bh2,
          bf16s* __restrict__ W1o, bf16s* __restrict__ W2o) {
    __shared__ __align__(16) bf16s lds_a[2][BM * LDK];
    __shared__ __align__(16) bf16s lds_b[2][BN * LDK];
    int tid = threadIdx.x, lane = tid & 31, wave = tid >> 5;
    int wm = wave & 3, wn = wave >> 2;
    long mBase = (long)blockIdx.x * BM;      // over BATCH*NTOK
    int  nBase = blockIdx.y * BN;            // over DPRM
    const bf16s* whT  = blockIdx.z ? whT2 : whT1;
    const float* bias = blockIdx.z ? bh2  : bh1;
    bf16s*       Wout = blockIdx.z ? W2o  : W1o;
    const bf16s* srcA = xp  + mBase * DIM;
    const bf16s* srcB = whT + (long)nBase * DIM;

    v8f acc[2][4]; zero_acc(acc);
#if HAS_TDM
    if (wave == 0) {
        tdm_load_tile(lds_a[0], srcA, DIM);
        tdm_load_tile(lds_b[0], srcB, DIM);
        __builtin_amdgcn_s_wait_tensorcnt(0);
    }
#else
    { RegsDirect ra, rb;
      ld_direct(ra, srcA, DIM, tid); ld_direct(rb, srcB, DIM, tid);
      st_direct(ra, lds_a[0], tid);  st_direct(rb, lds_b[0], tid); }
#endif
    __syncthreads();
    int cur = 0;
    for (int k = 0; k < DIM; k += BK) {
        int nxt = cur ^ 1;
        bool more = (k + BK) < DIM;
#if HAS_TDM
        if (wave == 0 && more) {
            tdm_load_tile(lds_a[nxt], srcA + k + BK, DIM);
            tdm_load_tile(lds_b[nxt], srcB + k + BK, DIM);
        }
        mma_step(lds_a[cur], lds_b[cur], wm, wn, lane, acc);
        if (wave == 0 && more) __builtin_amdgcn_s_wait_tensorcnt(0);
#else
        RegsDirect ra, rb;
        if (more) { ld_direct(ra, srcA + k + BK, DIM, tid);
                    ld_direct(rb, srcB + k + BK, DIM, tid); }
        mma_step(lds_a[cur], lds_b[cur], wm, wn, lane, acc);
        if (more) { st_direct(ra, lds_a[nxt], tid); st_direct(rb, lds_b[nxt], tid); }
#endif
        __syncthreads();
        cur = nxt;
    }
#pragma unroll
    for (int i = 0; i < 2; ++i)
#pragma unroll
        for (int j = 0; j < 4; ++j)
#pragma unroll
            for (int e = 0; e < 8; ++e) {
                int m  = wm * 32 + i * 16 + e + 8 * (lane >> 4);
                int nc = nBase + wn * 64 + j * 16 + (lane & 15);
                Wout[(mBase + m) * DPRM + nc] = f2bf(acc[i][j][e] + bias[nc]);
            }
}

// Kernel 4: Pm partials: Pm[z][b][p,d] = sum_{n in chunk z} W1[b,n,p]*Xp[b,n,d]
// Both operands need transpose -> manual register-split pipelined staging.
__global__ void __launch_bounds__(THREADS)
k_pm_partial(const bf16s* __restrict__ W1, const bf16s* __restrict__ xp,
             float* __restrict__ pmPart) {
    __shared__ __align__(16) bf16s lds_a[2][BM * LDK];
    __shared__ __align__(16) bf16s lds_b[2][BN * LDK];
    int tid = threadIdx.x, lane = tid & 31, wave = tid >> 5;
    int wm = wave & 3, wn = wave >> 2;
    int mb = blockIdx.x / (DIM / BN);    // p-tile 0..3
    int nb = blockIdx.x % (DIM / BN);    // d-tile 0..1
    int b  = blockIdx.y, z = blockIdx.z;
    int pBase = mb * BM, dBase = nb * BN;
    long nChunk = (long)z * (NTOK / KSPLIT);
    const bf16s* w1b = W1 + (long)b * NTOK * DPRM;
    const bf16s* xpb = xp + (long)b * NTOK * DIM;

    v8f acc[2][4]; zero_acc(acc);
    { RegsTrans ra, rb;
      ld_trans(ra, w1b + nChunk * DPRM + pBase, DPRM, tid);
      ld_trans(rb, xpb + nChunk * DIM  + dBase, DIM,  tid);
      st_trans(ra, lds_a[0], tid); st_trans(rb, lds_b[0], tid); }
    __syncthreads();
    int cur = 0;
    for (int k = 0; k < NTOK / KSPLIT; k += BK) {
        int nxt = cur ^ 1;
        bool more = (k + BK) < (NTOK / KSPLIT);
        RegsTrans ra, rb;
        if (more) {
            long nRow = nChunk + k + BK;
            ld_trans(ra, w1b + nRow * DPRM + pBase, DPRM, tid);
            ld_trans(rb, xpb + nRow * DIM  + dBase, DIM,  tid);
            __builtin_prefetch(w1b + (nRow + BK) * DPRM + pBase, 0, 0);
            __builtin_prefetch(xpb + (nRow + BK) * DIM  + dBase, 0, 0);
        }
        mma_step(lds_a[cur], lds_b[cur], wm, wn, lane, acc);
        if (more) { st_trans(ra, lds_a[nxt], tid); st_trans(rb, lds_b[nxt], tid); }
        __syncthreads();
        cur = nxt;
    }
    float* out = pmPart + ((long)z * BATCH + b) * DPRM * DIM;
#pragma unroll
    for (int i = 0; i < 2; ++i)
#pragma unroll
        for (int j = 0; j < 4; ++j)
#pragma unroll
            for (int e = 0; e < 8; ++e) {
                int p  = pBase + wm * 32 + i * 16 + e + 8 * (lane >> 4);
                int dc = dBase + wn * 64 + j * 16 + (lane & 15);
                out[(long)p * DIM + dc] = acc[i][j][e];
            }
}

// Kernel 5: reduce KSPLIT partials + exact-erf GELU -> bf16 A
__global__ void k_gelu(const float* __restrict__ pmPart, bf16s* __restrict__ Aact) {
    long i = (long)blockIdx.x * blockDim.x + threadIdx.x;   // over BATCH*DPRM*DIM
    const long tot = (long)BATCH * DPRM * DIM;
    if (i >= tot) return;
    float s = 0.f;
#pragma unroll
    for (int z = 0; z < KSPLIT; ++z) s += pmPart[(long)z * tot + i];
    float g = 0.5f * s * (1.0f + erff(s * 0.70710678118654752f));
    Aact[i] = f2bf(g);
}

// Kernel 6: Y[b][n,d] = W2[b][n,p] @ A[b][p,d]
// A-operand row-major -> TDM; B-operand transposed -> manual pipeline.
__global__ void __launch_bounds__(THREADS)
k_y(const bf16s* __restrict__ W2, const bf16s* __restrict__ Aact,
    float* __restrict__ Y) {
    __shared__ __align__(16) bf16s lds_a[2][BM * LDK];
    __shared__ __align__(16) bf16s lds_b[2][BN * LDK];
    int tid = threadIdx.x, lane = tid & 31, wave = tid >> 5;
    int wm = wave & 3, wn = wave >> 2;
    int mb = blockIdx.x / (DIM / BN);    // n-tile 0..31
    int nb = blockIdx.x % (DIM / BN);    // d-tile 0..1
    int b  = blockIdx.y;
    long mBase = (long)mb * BM;
    int  dBase = nb * BN;
    const bf16s* w2b = W2   + (long)b * NTOK * DPRM;
    const bf16s* ab  = Aact + (long)b * DPRM * DIM;
    const bf16s* srcA = w2b + mBase * DPRM;

    v8f acc[2][4]; zero_acc(acc);
#if HAS_TDM
    if (wave == 0) {
        tdm_load_tile(lds_a[0], srcA, DPRM);
        __builtin_amdgcn_s_wait_tensorcnt(0);
    }
#else
    { RegsDirect ra; ld_direct(ra, srcA, DPRM, tid); st_direct(ra, lds_a[0], tid); }
#endif
    { RegsTrans rb; ld_trans(rb, ab + dBase, DIM, tid); st_trans(rb, lds_b[0], tid); }
    __syncthreads();
    int cur = 0;
    for (int k = 0; k < DPRM; k += BK) {
        int nxt = cur ^ 1;
        bool more = (k + BK) < DPRM;
#if HAS_TDM
        if (wave == 0 && more) tdm_load_tile(lds_a[nxt], srcA + k + BK, DPRM);
        RegsTrans rb;
        if (more) ld_trans(rb, ab + (long)(k + BK) * DIM + dBase, DIM, tid);
        mma_step(lds_a[cur], lds_b[cur], wm, wn, lane, acc);
        if (more) st_trans(rb, lds_b[nxt], tid);
        if (wave == 0 && more) __builtin_amdgcn_s_wait_tensorcnt(0);
#else
        RegsDirect ra; RegsTrans rb;
        if (more) {
            ld_direct(ra, srcA + k + BK, DPRM, tid);
            ld_trans(rb, ab + (long)(k + BK) * DIM + dBase, DIM, tid);
        }
        mma_step(lds_a[cur], lds_b[cur], wm, wn, lane, acc);
        if (more) { st_direct(ra, lds_a[nxt], tid); st_trans(rb, lds_b[nxt], tid); }
#endif
        __syncthreads();
        cur = nxt;
    }
    float* yb = Y + (long)b * NTOK * DIM;
#pragma unroll
    for (int i = 0; i < 2; ++i)
#pragma unroll
        for (int j = 0; j < 4; ++j)
#pragma unroll
            for (int e = 0; e < 8; ++e) {
                int m  = (int)mBase + wm * 32 + i * 16 + e + 8 * (lane >> 4);
                int dc = dBase + wn * 64 + j * 16 + (lane & 15);
                yb[(long)m * DIM + dc] = acc[i][j][e];
            }
}

// ---------------------------------------------------------------------------
extern "C" void kernel_launch(void* const* d_in, const int* in_sizes, int n_in,
                              void* d_out, int out_size, void* d_ws, size_t ws_size,
                              hipStream_t stream) {
    const float* X   = (const float*)d_in[0];
    const float* P   = (const float*)d_in[1];
    const float* Wh1 = (const float*)d_in[2];
    const float* bh1 = (const float*)d_in[3];
    const float* Wh2 = (const float*)d_in[4];
    const float* bh2 = (const float*)d_in[5];
    float* Y = (float*)d_out;

    char* ws = (char*)d_ws;
    size_t off = 0;
    auto alloc = [&](size_t bytes) -> void* {
        void* p = ws + off;
        off += (bytes + 255) & ~((size_t)255);
        return p;
    };
    bf16s* xp     = (bf16s*)alloc((size_t)BATCH * NTOK * DIM * 2);
    bf16s* whT1   = (bf16s*)alloc((size_t)DPRM * DIM * 2);
    bf16s* whT2   = (bf16s*)alloc((size_t)DPRM * DIM * 2);
    bf16s* W1     = (bf16s*)alloc((size_t)BATCH * NTOK * DPRM * 2);
    bf16s* W2     = (bf16s*)alloc((size_t)BATCH * NTOK * DPRM * 2);
    float* pmPart = (float*)alloc((size_t)KSPLIT * BATCH * DPRM * DIM * 4);
    bf16s* Aact   = (bf16s*)alloc((size_t)BATCH * DPRM * DIM * 2);
    (void)ws_size; (void)in_sizes; (void)n_in; (void)out_size;

    long n4 = (long)BATCH * NTOK * DIM / 4;
    k_prep_xp<<<4096, 256, 0, stream>>>(X, P, xp, n4);
    k_prep_whT<<<(DIM * DPRM + 255) / 256, 256, 0, stream>>>(Wh1, Wh2, whT1, whT2);
    k_hyper_w<<<dim3(BATCH * NTOK / BM, DPRM / BN, 2), THREADS, 0, stream>>>(
        xp, whT1, whT2, bh1, bh2, W1, W2);
    k_pm_partial<<<dim3((DPRM / BM) * (DIM / BN), BATCH, KSPLIT), THREADS, 0, stream>>>(
        W1, xp, pmPart);
    long tot = (long)BATCH * DPRM * DIM;
    k_gelu<<<(int)((tot + 255) / 256), 256, 0, stream>>>(pmPart, Aact);
    k_y<<<dim3((NTOK / BM) * (DIM / BN), BATCH), THREADS, 0, stream>>>(W2, Aact, Y);
}
